// SwinTransformerBlock4D_49469433315672
// MI455X (gfx1250) — matrix operand order repaired and checked by reading, not compiled, not run on hardware
//
#include <hip/hip_runtime.h>
#include <math.h>

typedef __attribute__((ext_vector_type(16))) __bf16 v16bf;
typedef __attribute__((ext_vector_type(8)))  float  v8f;

union FragU {
  v16bf v;
  unsigned int u[8];
  unsigned short s[16];
};

// hardware f32 -> bf16 convert (fptrunc); backend emits v_cvt_*bf16*
__device__ __forceinline__ unsigned short f2bf(float f){
  union { __bf16 b; unsigned short s; } cv;
  cv.b = (__bf16)f;
  return cv.s;
}
// pack two f32 -> one dword of two bf16 (lets backend use packed cvt)
__device__ __forceinline__ unsigned int pkbf(float a, float b){
  union { __bf16 b[2]; unsigned int u; } cv;
  cv.b[0] = (__bf16)a; cv.b[1] = (__bf16)b;
  return cv.u;
}

__device__ __forceinline__ v8f wmma_bf16(v16bf a, v16bf b, v8f c){
  return __builtin_amdgcn_wmma_f32_16x16x32_bf16(false, a, false, b, (short)0, c, false, false);
}

// ---------------------------------------------------------------------------
// Kernel 0: pack all weights into WMMA fragment layouts (bf16) in workspace.
//   frag id f (global, 1KB each = 32 lanes * 32B):
//     [  0, 96): Wqkv  B-frags  (K=128 -> ks 0..3, N=384 -> nt 0..23)
//     [ 96,128): Wproj B-frags  (K=128, N=128)
//     [128,256): Wfc1  A-frags  (M=512 -> mt 0..31, K=128 -> ks 0..3)
//     [256,384): Wfc2  A-frags  (M=128 -> mt 0..7,  K=512 -> ks 0..15)
// ---------------------------------------------------------------------------
__global__ void pack_weights_kernel(const float* __restrict__ wqkv,
                                    const float* __restrict__ wproj,
                                    const float* __restrict__ wfc1,
                                    const float* __restrict__ wfc2,
                                    __bf16* __restrict__ ws)
{
  int gw = (blockIdx.x * blockDim.x + threadIdx.x) >> 5;  // fragment id
  int lane = threadIdx.x & 31;
  if (gw >= 384) return;
  int hi = lane >> 4, l15 = lane & 15;

  const float* src; int row, rs, kbase; bool aType;
  if (gw < 96)       { int nt=gw>>2,        ks=gw&3;       src=wqkv;  row=nt*16+l15; rs=128; kbase=ks*32; aType=false; }
  else if (gw < 128) { int j=gw-96, nt=j>>2, ks=j&3;       src=wproj; row=nt*16+l15; rs=128; kbase=ks*32; aType=false; }
  else if (gw < 256) { int j=gw-128, mt=j>>2, ks=j&3;      src=wfc1;  row=mt*16+l15; rs=128; kbase=ks*32; aType=true;  }
  else               { int j=gw-256, mt=j>>4, ks=j&15;     src=wfc2;  row=mt*16+l15; rs=512; kbase=ks*32; aType=true;  }

  FragU f;
  #pragma unroll
  for (int e = 0; e < 16; ++e){
    int k = aType ? (kbase + (e<8 ? e : e+8) + (hi ? 8 : 0))
                  : (kbase + (e<8 ? e : e+8) + (hi ? 8 : 0));
    f.s[e] = f2bf(src[row*rs + k]);
  }
  uint4* dst = (uint4*)(ws + (size_t)gw*512 + lane*16);
  dst[0] = make_uint4(f.u[0], f.u[1], f.u[2], f.u[3]);
  dst[1] = make_uint4(f.u[4], f.u[5], f.u[6], f.u[7]);
}

// ---------------------------------------------------------------------------
// Kernel 1: fused LN1 + roll/partition + QKV + masked attention + proj + res.
// One block per window (512 blocks, 256 threads = 8 waves).
// ---------------------------------------------------------------------------
__global__ __launch_bounds__(256, 1) void swin_attn_kernel(
    const float* __restrict__ x,
    const float* __restrict__ g1, const float* __restrict__ b1,
    const float* __restrict__ bqkv, const float* __restrict__ bproj,
    const __bf16* __restrict__ wqkvf, const __bf16* __restrict__ wprojf,
    float* __restrict__ xbuf)
{
  __shared__ __align__(32) __bf16 a_lds[16*4*32*16];   // LN out as A-frags, 64KB
  __shared__ __align__(16) __bf16 qk_lds[256*264];     // [token][chan 0..255] Q|K
  __shared__ __align__(16) __bf16 v_lds[128*264];      // [chan][token] V
  const int QKP = 264, VP = 264;
  const v8f vzero = {0.f,0.f,0.f,0.f,0.f,0.f,0.f,0.f};

  const int tid = threadIdx.x, lane = tid & 31, wv = tid >> 5;
  const int l15 = lane & 15, hi = lane >> 4;
  const int wid = blockIdx.x, bb = wid >> 8, nw = wid & 255;
  const int wd=(nw>>6)&3, wh=(nw>>4)&3, wwc=(nw>>2)&3, wtc=nw&3;

  // ---- phase 1: gather (roll+partition) + LayerNorm -> A fragments ----
  float4 gg = *(const float4*)(g1 + 4*lane);
  float4 gb = *(const float4*)(b1 + 4*lane);
  for (int i = 0; i < 32; ++i){
    int t = wv*32 + i;
    int td=(t>>6)&3, th=(t>>4)&3, tw=(t>>2)&3, tt=t&3;
    int qd=(wd*4+td+2)&15, qh=(wh*4+th+2)&15, qw=(wwc*4+tw+2)&15, qv=(wtc*4+tt+2)&15;
    int g = (((bb*16+qd)*16+qh)*16+qw)*16+qv;
    float4 xv = *(const float4*)(x + (size_t)g*128 + 4*lane);
    float s = xv.x+xv.y+xv.z+xv.w;
    #pragma unroll
    for (int o=16;o>=1;o>>=1) s += __shfl_xor(s,o);
    float mean = s*(1.f/128.f);
    float d0=xv.x-mean, d1=xv.y-mean, d2=xv.z-mean, d3=xv.w-mean;
    float vs = d0*d0+d1*d1+d2*d2+d3*d3;
    #pragma unroll
    for (int o=16;o>=1;o>>=1) vs += __shfl_xor(vs,o);
    float rstd = rsqrtf(vs*(1.f/128.f) + 1e-5f);
    float y0=d0*rstd*gg.x+gb.x, y1=d1*rstd*gg.y+gb.y;
    float y2=d2*rstd*gg.z+gb.z, y3=d3*rstd*gg.w+gb.w;
    int mt = t>>4, mr = t&15;
    #pragma unroll
    for (int p=0;p<2;p++){
      int cc = 4*lane + 2*p;
      int ks = cc>>5, k5 = cc&31;
      int half = (k5>>3)&1;
      int e = (k5&7) + ((k5>>4)<<3);
      int fl = mr + half*16;
      unsigned int pk = p ? pkbf(y2,y3) : pkbf(y0,y1);
      *(unsigned int*)&a_lds[(((mt*4+ks)*32)+fl)*16 + e] = pk;
    }
  }
  __syncthreads();

  // ---- phase 2: QKV GEMM (256x384 = 16 mtiles x 24 ntiles) ----
  {
    const v16bf* Af = (const v16bf*)a_lds;
    const v16bf* Bf = (const v16bf*)wqkvf;
    for (int tile = wv; tile < 384; tile += 8){
      int nt = tile % 24, mt = tile / 24;
      v8f acc = vzero;
      #pragma unroll
      for (int ks=0; ks<4; ++ks){
        v16bf a = Af[(mt*4+ks)*32 + lane];
        v16bf b = Bf[(nt*4+ks)*32 + lane];
        acc = wmma_bf16(a, b, acc);
      }
      int col = nt*16 + l15;
      float bias = bqkv[col];
      int tokb = mt*16 + hi*8;
      if (nt < 16){                      // Q (scaled) and K -> [token][chan]
        float sc = (nt < 8) ? 0.17677669529663689f : 1.0f;
        #pragma unroll
        for (int r=0;r<8;r++)
          *(unsigned short*)&qk_lds[(tokb+r)*QKP + col] = f2bf((acc[r]+bias)*sc);
      } else {                           // V -> [chan][token] (transposed)
        int ch = (nt-16)*16 + l15;
        unsigned int u0 = pkbf(acc[0]+bias, acc[1]+bias);
        unsigned int u1 = pkbf(acc[2]+bias, acc[3]+bias);
        unsigned int u2 = pkbf(acc[4]+bias, acc[5]+bias);
        unsigned int u3 = pkbf(acc[6]+bias, acc[7]+bias);
        *(uint4*)&v_lds[ch*VP + tokb] = make_uint4(u0,u1,u2,u3);
      }
    }
  }
  __syncthreads();

  // ---- phase 3: S^T = K*Q^T, analytic mask, softmax, O^T = V^T*P^T, proj ----
  {
    const v16bf* Wp = (const v16bf*)wprojf;
    int m3 = ((wd==3)?8:0)|((wh==3)?4:0)|((wwc==3)?2:0)|((wtc==3)?1:0);
    for (int qi=0; qi<2; ++qi){
      int qt = wv*2 + qi;
      int tq = qt*16 + l15;                      // this lane's query token
      unsigned rq = 0;
      if (m3){
        if (m3&8) rq |= (1u + ((tq>>7)&1u)) << 6;
        if (m3&4) rq |= (1u + ((tq>>5)&1u)) << 4;
        if (m3&2) rq |= (1u + ((tq>>3)&1u)) << 2;
        if (m3&1) rq |= (1u + ((tq>>1)&1u));
      }
      v8f pacc[8];
      #pragma unroll
      for (int n2=0;n2<8;n2++) pacc[n2] = vzero;

      for (int h=0; h<4; ++h){
        FragU qf;                                // B-frag of Q^T
        {
          const __bf16* qp = qk_lds + tq*QKP + h*32 + hi*8;
          #pragma unroll
          for (int i2=0;i2<8;i2++) qf.u[i2] = *(const unsigned int*)(qp + (i2<4 ? 2*i2 : 2*i2+8));
        }
        v8f st[16];                              // S^T column block (256 kv x 16 q)
        #pragma unroll
        for (int kt=0;kt<16;kt++){
          FragU kf;                              // A-frag of K
          const __bf16* kp = qk_lds + (kt*16 + l15)*QKP + 128 + h*32 + hi*8;
          #pragma unroll
          for (int i2=0;i2<8;i2++) kf.u[i2] = *(const unsigned int*)(kp + (i2<4 ? 2*i2 : 2*i2+8));
          st[kt] = wmma_bf16(kf.v, qf.v, vzero);
        }
        if (m3){                                  // analytic Swin shift mask
          #pragma unroll
          for (int kt=0;kt<16;kt++){
            #pragma unroll
            for (int r=0;r<8;r++){
              int kv = kt*16 + r + hi*8;
              unsigned rk = 0;
              if (m3&8) rk |= (1u + ((kv>>7)&1u)) << 6;
              if (m3&4) rk |= (1u + ((kv>>5)&1u)) << 4;
              if (m3&2) rk |= (1u + ((kv>>3)&1u)) << 2;
              if (m3&1) rk |= (1u + ((kv>>1)&1u));
              if (rk != rq) st[kt][r] += -100.0f;
            }
          }
        }
        float mx = -3.0e38f;
        #pragma unroll
        for (int kt=0;kt<16;kt++){
          #pragma unroll
          for (int r=0;r<8;r++) mx = fmaxf(mx, st[kt][r]);
        }
        mx = fmaxf(mx, __shfl_xor(mx,16));
        float sum = 0.f;
        #pragma unroll
        for (int kt=0;kt<16;kt++){
          #pragma unroll
          for (int r=0;r<8;r++){ float p = __expf(st[kt][r]-mx); st[kt][r] = p; sum += p; }
        }
        sum += __shfl_xor(sum,16);
        float rs = 1.0f/sum;

        v8f o0 = vzero, o1 = vzero;              // O^T accumulators (32 chans)
        #pragma unroll
        for (int kv2=0;kv2<8;kv2++){
          // B-frag of P^T: convert first, pack rows (2r,2r+1), C-layout already matches B-frag; shfl results unused
          FragU pf;
          #pragma unroll
          for (int rr=0;rr<4;rr++){
            unsigned int pa = pkbf(st[2*kv2  ][2*rr], st[2*kv2  ][2*rr+1]);
            unsigned int pb = pkbf(st[2*kv2+1][2*rr], st[2*kv2+1][2*rr+1]);
            unsigned int pas = __shfl_xor(pa,16);
            unsigned int pbs = __shfl_xor(pb,16);
            pf.u[rr]   = pa;
            pf.u[rr+4] = pb;
          }
          FragU vf0, vf1;                        // A-frags of V^T
          const __bf16* vp0 = v_lds + (h*32 + l15)*VP + kv2*32 + hi*8;
          const __bf16* vp1 = vp0 + 16*VP;
          #pragma unroll
          for (int i2=0;i2<8;i2++){
            int off = (i2<4 ? 2*i2 : 2*i2+8);
            vf0.u[i2] = *(const unsigned int*)(vp0 + off);
            vf1.u[i2] = *(const unsigned int*)(vp1 + off);
          }
          o0 = wmma_bf16(vf0.v, pf.v, o0);
          o1 = wmma_bf16(vf1.v, pf.v, o1);
        }
        FragU of;                                // O^T C-layout == O A-frag layout
        #pragma unroll
        for (int i2=0;i2<4;i2++){
          of.u[i2]   = pkbf(o0[2*i2]*rs, o0[2*i2+1]*rs);
          of.u[i2+4] = pkbf(o1[2*i2]*rs, o1[2*i2+1]*rs);
        }
        #pragma unroll
        for (int n2=0;n2<8;n2++){
          v16bf wb = Wp[(n2*4 + h)*32 + lane];
          pacc[n2] = wmma_bf16(of.v, wb, pacc[n2]);
        }
      }
      // epilogue: bias + shortcut, store residual x to xbuf (reverse+roll map)
      int gg8[8];
      #pragma unroll
      for (int r=0;r<8;r++){
        int t = qt*16 + r + hi*8;
        int td=(t>>6)&3, th=(t>>4)&3, tw=(t>>2)&3, tt=t&3;
        int qd=(wd*4+td+2)&15, qh=(wh*4+th+2)&15, qw=(wwc*4+tw+2)&15, qv=(wtc*4+tt+2)&15;
        gg8[r] = ((((bb*16+qd)*16+qh)*16+qw)*16+qv)*128;
      }
      #pragma unroll
      for (int n2=0;n2<8;n2++){
        int col = n2*16 + l15;
        float bp = bproj[col];
        #pragma unroll
        for (int r=0;r<8;r++)
          xbuf[gg8[r] + col] = pacc[n2][r] + bp + x[gg8[r] + col];
      }
    }
  }
}

// ---------------------------------------------------------------------------
// Kernel 2: fused LN2 + FC1 + exact GELU + FC2 + residual.
// 512 blocks x 256 threads, 256 tokens per block.
// Computes H^T = W1 * X^T and OUT^T = W2 * gelu(H^T) (C-layout reused as B-frag).
// ---------------------------------------------------------------------------
__global__ __launch_bounds__(256, 1) void swin_mlp_kernel(
    const float* __restrict__ xbuf,
    const float* __restrict__ g2, const float* __restrict__ b2,
    const __bf16* __restrict__ w1f, const __bf16* __restrict__ w2f,
    const float* __restrict__ bfc1, const float* __restrict__ bfc2,
    float* __restrict__ out)
{
  __shared__ __align__(32) __bf16 x_lds[256*136];
  __shared__ __align__(16) float  o_lds[256*132];
  const int XP = 136, OP = 132;
  const v8f vzero = {0.f,0.f,0.f,0.f,0.f,0.f,0.f,0.f};

  const int tid = threadIdx.x, lane = tid & 31, wv = tid >> 5;
  const int l15 = lane & 15, hi = lane >> 4;
  const size_t base = (size_t)blockIdx.x * 256;

  float4 gg = *(const float4*)(g2 + 4*lane);
  float4 gb = *(const float4*)(b2 + 4*lane);
  for (int i = 0; i < 32; ++i){
    int tl = wv*32 + i;
    float4 xv = *(const float4*)(xbuf + (base+tl)*128 + 4*lane);
    float s = xv.x+xv.y+xv.z+xv.w;
    #pragma unroll
    for (int o=16;o>=1;o>>=1) s += __shfl_xor(s,o);
    float mean = s*(1.f/128.f);
    float d0=xv.x-mean, d1=xv.y-mean, d2=xv.z-mean, d3=xv.w-mean;
    float vs = d0*d0+d1*d1+d2*d2+d3*d3;
    #pragma unroll
    for (int o=16;o>=1;o>>=1) vs += __shfl_xor(vs,o);
    float rstd = rsqrtf(vs*(1.f/128.f) + 1e-5f);
    unsigned int p0 = pkbf(d0*rstd*gg.x+gb.x, d1*rstd*gg.y+gb.y);
    unsigned int p1 = pkbf(d2*rstd*gg.z+gb.z, d3*rstd*gg.w+gb.w);
    unsigned int* dst = (unsigned int*)&x_lds[tl*XP + 4*lane];
    dst[0] = p0; dst[1] = p1;
  }
  __syncthreads();

  const v16bf* W1 = (const v16bf*)w1f;
  const v16bf* W2 = (const v16bf*)w2f;
  for (int nq=0; nq<2; ++nq){
    int nt = wv*2 + nq;                          // token ntile
    FragU xf[4];                                 // B-frags of X^T, reused 16x
    #pragma unroll
    for (int c=0;c<4;c++){
      const __bf16* xp = x_lds + (nt*16 + l15)*XP + c*32 + hi*8;
      #pragma unroll
      for (int i2=0;i2<8;i2++) xf[c].u[i2] = *(const unsigned int*)(xp + (i2<4 ? 2*i2 : 2*i2+8));
    }
    v8f acc2[8];
    #pragma unroll
    for (int om=0;om<8;om++) acc2[om] = vzero;

    for (int kk=0; kk<16; ++kk){                 // 32 hidden chans per step
      v8f h0 = vzero, h1 = vzero;
      #pragma unroll
      for (int c=0;c<4;c++) h0 = wmma_bf16(W1[((2*kk  )*4 + c)*32 + lane], xf[c].v, h0);
      #pragma unroll
      for (int c=0;c<4;c++) h1 = wmma_bf16(W1[((2*kk+1)*4 + c)*32 + lane], xf[c].v, h1);
      #pragma unroll
      for (int r=0;r<8;r++){
        float u0 = h0[r] + bfc1[kk*32 + r + hi*8];
        float u1 = h1[r] + bfc1[kk*32 + 16 + r + hi*8];
        h0[r] = 0.5f*u0*(1.f + erff(u0*0.70710678118654752f));
        h1[r] = 0.5f*u1*(1.f + erff(u1*0.70710678118654752f));
      }
      // B-frag of H^T: convert, pack row pairs, C-layout already matches B-frag; shfl results unused
      FragU hf;
      #pragma unroll
      for (int rr=0;rr<4;rr++){
        unsigned int pa = pkbf(h0[2*rr], h0[2*rr+1]);
        unsigned int pb = pkbf(h1[2*rr], h1[2*rr+1]);
        unsigned int pas = __shfl_xor(pa,16);
        unsigned int pbs = __shfl_xor(pb,16);
        hf.u[rr]   = pa;
        hf.u[rr+4] = pb;
      }
      #pragma unroll
      for (int om=0;om<8;om++)
        acc2[om] = wmma_bf16(W2[(om*16 + kk)*32 + lane], hf.v, acc2[om]);
    }
    #pragma unroll
    for (int om=0;om<8;om++){
      #pragma unroll
      for (int r=0;r<8;r++)
        o_lds[(nt*16 + l15)*OP + om*16 + r + hi*8] = acc2[om][r] + bfc2[om*16 + r + hi*8];
    }
  }
  __syncthreads();

  for (int i = 0; i < 32; ++i){
    int tl = wv*32 + i;
    float4 xv = *(const float4*)(xbuf + (base+tl)*128 + 4*lane);
    float4 ov = *(const float4*)&o_lds[tl*OP + 4*lane];
    float4 r; r.x=xv.x+ov.x; r.y=xv.y+ov.y; r.z=xv.z+ov.z; r.w=xv.w+ov.w;
    *(float4*)(out + (base+tl)*128 + 4*lane) = r;
  }
}

// ---------------------------------------------------------------------------
extern "C" void kernel_launch(void* const* d_in, const int* in_sizes, int n_in,
                              void* d_out, int out_size, void* d_ws, size_t ws_size,
                              hipStream_t stream)
{
  (void)in_sizes; (void)n_in; (void)out_size; (void)ws_size;
  const float* x     = (const float*)d_in[0];
  // d_in[1] = mask_matrix: not read (mask computed analytically)
  const float* g1    = (const float*)d_in[2];
  const float* b1    = (const float*)d_in[3];
  const float* wqkv  = (const float*)d_in[4];
  const float* bqkv  = (const float*)d_in[5];
  const float* wproj = (const float*)d_in[6];
  const float* bproj = (const float*)d_in[7];
  const float* g2    = (const float*)d_in[8];
  const float* b2    = (const float*)d_in[9];
  const float* wfc1  = (const float*)d_in[10];
  const float* bfc1  = (const float*)d_in[11];
  const float* wfc2  = (const float*)d_in[12];
  const float* bfc2  = (const float*)d_in[13];
  float* out  = (float*)d_out;

  __bf16* ws  = (__bf16*)d_ws;                       // 384 KB of packed fragments
  float*  xbuf = (float*)((char*)d_ws + 384*1024);   // 64 MB residual buffer

  pack_weights_kernel<<<48, 256, 0, stream>>>(wqkv, wproj, wfc1, wfc2, ws);
  swin_attn_kernel<<<512, 256, 0, stream>>>(x, g1, b1, bqkv, bproj,
                                            ws, ws + 96*512, xbuf);
  swin_mlp_kernel<<<512, 256, 0, stream>>>(xbuf, g2, b2,
                                           ws + 128*512, ws + 256*512,
                                           bfc1, bfc2, out);
}
